// MoleRecLayer_35553739276801
// MI455X (gfx1250) — matrix operand order repaired and checked by reading, not compiled, not run on hardware
//
#include <hip/hip_runtime.h>
#include <hip/hip_bf16.h>
#include <math.h>

typedef __attribute__((ext_vector_type(8)))  float          v8f;
typedef __attribute__((ext_vector_type(16))) __bf16         v16bf;
typedef __attribute__((ext_vector_type(16))) unsigned short u16x16;

#define BN_INV 0.99999500003749969f   // 1/sqrt(1+1e-5)
#define NEGV   (-4294967296.0f)       // -(1<<32)

// ---------------------------------------------------------------- helpers

__device__ __forceinline__ v8f v8f_zero(){
  v8f z = {0.f,0.f,0.f,0.f,0.f,0.f,0.f,0.f};
  return z;
}

__device__ __forceinline__ v16bf bf_zero(){
  v16bf o;
#pragma unroll
  for (int j = 0; j < 16; ++j) o[j] = (__bf16)0.0f;
  return o;
}

// convert two float4 runs into 8 consecutive bf16 vector elements
__device__ __forceinline__ void cvt8(v16bf& o, int base, float4 a, float4 b){
  o[base+0]=(__bf16)a.x; o[base+1]=(__bf16)a.y; o[base+2]=(__bf16)a.z; o[base+3]=(__bf16)a.w;
  o[base+4]=(__bf16)b.x; o[base+5]=(__bf16)b.y; o[base+6]=(__bf16)b.z; o[base+7]=(__bf16)b.w;
}

// base pointer to a lane's slice of packed fragment `frag`; neighbouring
// fragments are +32 u16x16 elements (1 KB) away -> immediate offsets.
__device__ __forceinline__ const u16x16* bfrag_base(const unsigned short* __restrict__ P,
                                                    int frag, int lane){
  return (const u16x16*)(P + ((size_t)frag << 9) + (lane << 4));
}

__device__ __forceinline__ v16bf asbf(u16x16 u){
  union { u16x16 u; v16bf b; } fu;
  fu.u = u;
  return fu.b;
}

// A fragment, branch-free fast path: rows valid, K-chunk fully in-bounds.
__device__ __forceinline__ v16bf afrag_fast(const float* __restrict__ A, int mbase,
                                            int K, int kbase, int lane){
  int m  = mbase + (lane & 15);
  int hi = lane >> 4;
  const float* row = A + (size_t)m * (size_t)K + kbase + (hi << 3);
  float4 p0 = *(const float4*)(row);
  float4 p1 = *(const float4*)(row + 4);
  float4 p2 = *(const float4*)(row + 16);
  float4 p3 = *(const float4*)(row + 20);
  v16bf o;
  cvt8(o, 0, p0, p1);
  cvt8(o, 8, p2, p3);
  return o;
}

// A fragment, guarded (edge tiles / K tail only)
__device__ __forceinline__ v16bf afrag_guard(const float* __restrict__ A, int mbase,
                                             int M, int K, int kbase, int lane){
  int m  = mbase + (lane & 15);
  int hi = lane >> 4;
  int k0 = kbase + (hi << 3);
  int k1 = k0 + 16;
  bool vr = (m < M);
  const float* row = A + (size_t)m * (size_t)K;
  v16bf o;
#pragma unroll
  for (int j = 0; j < 8; ++j){
    int k = k0 + j;
    o[j] = (__bf16)((vr && k < K) ? row[k] : 0.0f);
  }
#pragma unroll
  for (int j = 0; j < 8; ++j){
    int k = k1 + j;
    o[8+j] = (__bf16)((vr && k < K) ? row[k] : 0.0f);
  }
  return o;
}

// ------------------------------------------------------- weight packing
// B (fp32 [K,N] row-major) -> bf16 fragments, kc-major: frag = kc*nt_cnt + nt
// elem = lane*16 + j : n = nt*16 + lane%16, k = kc*32 + (j&7) + (lane>=16)*8 + (j>=8)*16
__global__ __launch_bounds__(256) void pack_b_kernel(const float* __restrict__ B,
                                                     unsigned short* __restrict__ P,
                                                     int K, int N, int nt_cnt, int total){
  int i = blockIdx.x * 256 + threadIdx.x;
  if (i >= total) return;
  int f    = i >> 9;
  int r    = i & 511;
  int lane = r >> 4;
  int j    = r & 15;
  int kc   = f / nt_cnt;
  int nt   = f - kc * nt_cnt;
  int n    = (nt << 4) + (lane & 15);
  int k    = (kc << 5) + (j & 7) + ((lane >> 4) << 3) + ((j >> 3) << 4);
  float v  = (k < K && n < N) ? B[(size_t)k * N + n] : 0.0f;
  __bf16 h = (__bf16)v;
  P[i] = *(unsigned short*)&h;
}

// ------------------------------------------------------------ generic GEMM
// C[M,N] = A[M,K] @ Bpacked + bias ; optional relu.
// Wave computes 16x64 strip; 4 contiguous B frags per K-chunk share one base.
__global__ __launch_bounds__(256) void gemm_bf16_kernel(
    const float* __restrict__ A, const unsigned short* __restrict__ Bp,
    const float* __restrict__ bias, float* __restrict__ C,
    int M, int N, int K, int kc_cnt, int relu)
{
  int lane = threadIdx.x & 31;
  int wid  = threadIdx.x >> 5;
  int nt_cnt = N >> 4;
  int ng     = N >> 6;
  int mt_cnt = (M + 15) >> 4;
  int tile = blockIdx.x * 8 + wid;
  if (tile >= mt_cnt * ng) return;
  int mt  = tile / ng;
  int nt0 = (tile - mt * ng) << 2;
  int mb  = mt << 4;
  bool mfull  = (mb + 16) <= M;
  int kc_full = K >> 5;

  v8f acc0 = v8f_zero(), acc1 = v8f_zero(), acc2 = v8f_zero(), acc3 = v8f_zero();

  if (mfull){
    for (int kc = 0; kc < kc_full; ++kc){
      const u16x16* bp = bfrag_base(Bp, kc * nt_cnt + nt0, lane);
      u16x16 u0 = bp[0], u1 = bp[32], u2 = bp[64], u3 = bp[96];
      v16bf a  = afrag_fast(A, mb, K, kc << 5, lane);
      acc0 = __builtin_amdgcn_wmma_f32_16x16x32_bf16(false, a, false, asbf(u0), (short)0, acc0, false, false);
      acc1 = __builtin_amdgcn_wmma_f32_16x16x32_bf16(false, a, false, asbf(u1), (short)0, acc1, false, false);
      acc2 = __builtin_amdgcn_wmma_f32_16x16x32_bf16(false, a, false, asbf(u2), (short)0, acc2, false, false);
      acc3 = __builtin_amdgcn_wmma_f32_16x16x32_bf16(false, a, false, asbf(u3), (short)0, acc3, false, false);
    }
    if (kc_full < kc_cnt){
      const u16x16* bp = bfrag_base(Bp, kc_full * nt_cnt + nt0, lane);
      u16x16 u0 = bp[0], u1 = bp[32], u2 = bp[64], u3 = bp[96];
      v16bf a  = afrag_guard(A, mb, M, K, kc_full << 5, lane);
      acc0 = __builtin_amdgcn_wmma_f32_16x16x32_bf16(false, a, false, asbf(u0), (short)0, acc0, false, false);
      acc1 = __builtin_amdgcn_wmma_f32_16x16x32_bf16(false, a, false, asbf(u1), (short)0, acc1, false, false);
      acc2 = __builtin_amdgcn_wmma_f32_16x16x32_bf16(false, a, false, asbf(u2), (short)0, acc2, false, false);
      acc3 = __builtin_amdgcn_wmma_f32_16x16x32_bf16(false, a, false, asbf(u3), (short)0, acc3, false, false);
    }
  } else {
    for (int kc = 0; kc < kc_cnt; ++kc){
      const u16x16* bp = bfrag_base(Bp, kc * nt_cnt + nt0, lane);
      u16x16 u0 = bp[0], u1 = bp[32], u2 = bp[64], u3 = bp[96];
      v16bf a  = afrag_guard(A, mb, M, K, kc << 5, lane);
      acc0 = __builtin_amdgcn_wmma_f32_16x16x32_bf16(false, a, false, asbf(u0), (short)0, acc0, false, false);
      acc1 = __builtin_amdgcn_wmma_f32_16x16x32_bf16(false, a, false, asbf(u1), (short)0, acc1, false, false);
      acc2 = __builtin_amdgcn_wmma_f32_16x16x32_bf16(false, a, false, asbf(u2), (short)0, acc2, false, false);
      acc3 = __builtin_amdgcn_wmma_f32_16x16x32_bf16(false, a, false, asbf(u3), (short)0, acc3, false, false);
    }
  }

  int hi = lane >> 4;
  int nl = lane & 15;
  v8f* accs[4] = {&acc0, &acc1, &acc2, &acc3};
#pragma unroll
  for (int tt = 0; tt < 4; ++tt){
    int n = ((nt0 + tt) << 4) + nl;
    float bs = bias ? bias[n] : 0.0f;
    v8f& acc = *accs[tt];
    if (mfull){
#pragma unroll
      for (int d = 0; d < 8; ++d){
        int m = mb + d + (hi ? 8 : 0);
        float v = acc[d] + bs;
        if (relu) v = v > 0.0f ? v : 0.0f;
        C[(size_t)m * N + n] = v;
      }
    } else {
#pragma unroll
      for (int d = 0; d < 8; ++d){
        int m = mb + d + (hi ? 8 : 0);
        if (m < M){
          float v = acc[d] + bs;
          if (relu) v = v > 0.0f ? v : 0.0f;
          C[(size_t)m * N + n] = v;
        }
      }
    }
  }
}

// ------------------------------------------------------------ GIN kernels

__global__ __launch_bounds__(256) void node_embed_kernel(
    const int* __restrict__ x, const float* __restrict__ emb,
    float* __restrict__ h, int N)
{
  long long gid = (long long)blockIdx.x * 256 + threadIdx.x;
  int n = (int)(gid >> 5);
  if (n >= N) return;
  int c = (int)(gid & 31) * 4;
  float a0=0.f,a1=0.f,a2=0.f,a3=0.f;
#pragma unroll
  for (int f = 0; f < 9; ++f){
    int id = x[(size_t)n * 9 + f];
    const float* r = emb + (size_t)(f * 64 + id) * 128 + c;
    a0 += r[0]; a1 += r[1]; a2 += r[2]; a3 += r[3];
  }
  float* d = h + (size_t)n * 128 + c;
  d[0]=a0; d[1]=a1; d[2]=a2; d[3]=a3;
}

__global__ __launch_bounds__(256) void edge_msg_kernel(
    const int* __restrict__ eidx, const int* __restrict__ eattr,
    const float* __restrict__ bemb, const float* __restrict__ h,
    float* __restrict__ agg, int E)
{
  long long gid = (long long)blockIdx.x * 256 + threadIdx.x;
  int e = (int)(gid >> 5);
  if (e >= E) return;
  int c = (int)(gid & 31) * 4;
  int tgt = eidx[e];
  int src = eidx[(size_t)E + e];
  const float* hs = h + (size_t)src * 128 + c;
  float a0 = hs[0], a1 = hs[1], a2 = hs[2], a3 = hs[3];
#pragma unroll
  for (int f = 0; f < 3; ++f){
    int id = eattr[(size_t)e * 3 + f];
    const float* r = bemb + (size_t)(f * 64 + id) * 128 + c;
    a0 += r[0]; a1 += r[1]; a2 += r[2]; a3 += r[3];
  }
  a0 = a0 > 0.f ? a0 : 0.f;  a1 = a1 > 0.f ? a1 : 0.f;
  a2 = a2 > 0.f ? a2 : 0.f;  a3 = a3 > 0.f ? a3 : 0.f;
  float* dst = agg + (size_t)tgt * 128 + c;
  atomicAdd(dst + 0, a0); atomicAdd(dst + 1, a1);
  atomicAdd(dst + 2, a2); atomicAdd(dst + 3, a3);
}

__global__ __launch_bounds__(64) void gin_mlp_kernel(
    float* __restrict__ h, const float* __restrict__ agg,
    const unsigned short* __restrict__ w1p, const unsigned short* __restrict__ w2p,
    const float* __restrict__ b1, const float* __restrict__ g1, const float* __restrict__ be1,
    const float* __restrict__ b2, const float* __restrict__ g2, const float* __restrict__ be2,
    const float* __restrict__ epsp, int N, int lastlayer)
{
  __shared__ float tbuf[2][16 * 260];
  int lane = threadIdx.x & 31;
  int wid  = threadIdx.x >> 5;
  int hi   = lane >> 4;
  int m    = lane & 15;
  int nb   = blockIdx.x * 2 + wid;
  if (nb * 16 >= N) return;
  int node = nb * 16 + m;
  bool vrow = node < N;
  float ep1 = 1.0f + epsp[0];
  const float* hrow = h   + (size_t)node * 128;
  const float* arow = agg + (size_t)node * 128;

  v16bf af1[4];
#pragma unroll
  for (int kc = 0; kc < 4; ++kc){
    if (vrow){
      int k0 = (kc << 5) + (hi << 3);
      int k1 = k0 + 16;
      float4 h0 = *(const float4*)(hrow + k0);
      float4 h1 = *(const float4*)(hrow + k0 + 4);
      float4 a0 = *(const float4*)(arow + k0);
      float4 a1 = *(const float4*)(arow + k0 + 4);
      float4 h2 = *(const float4*)(hrow + k1);
      float4 h3 = *(const float4*)(hrow + k1 + 4);
      float4 a2 = *(const float4*)(arow + k1);
      float4 a3 = *(const float4*)(arow + k1 + 4);
      float4 z0, z1, z2, z3;
      z0.x = ep1*h0.x + a0.x; z0.y = ep1*h0.y + a0.y; z0.z = ep1*h0.z + a0.z; z0.w = ep1*h0.w + a0.w;
      z1.x = ep1*h1.x + a1.x; z1.y = ep1*h1.y + a1.y; z1.z = ep1*h1.z + a1.z; z1.w = ep1*h1.w + a1.w;
      z2.x = ep1*h2.x + a2.x; z2.y = ep1*h2.y + a2.y; z2.z = ep1*h2.z + a2.z; z2.w = ep1*h2.w + a2.w;
      z3.x = ep1*h3.x + a3.x; z3.y = ep1*h3.y + a3.y; z3.z = ep1*h3.z + a3.z; z3.w = ep1*h3.w + a3.w;
      cvt8(af1[kc], 0, z0, z1);
      cvt8(af1[kc], 8, z2, z3);
    } else {
      af1[kc] = bf_zero();
    }
  }

  float* t = &tbuf[wid][0];
  // matmul1: [16x128] @ [128x256]  (w1 packed kc-major, nt_cnt = 16)
#pragma unroll
  for (int nt = 0; nt < 16; ++nt){
    const u16x16* bp = bfrag_base(w1p, nt, lane);     // kc stride = 16 frags
    u16x16 u0 = bp[0], u1 = bp[512], u2 = bp[1024], u3 = bp[1536];
    v8f acc = v8f_zero();
    acc = __builtin_amdgcn_wmma_f32_16x16x32_bf16(false, af1[0], false, asbf(u0), (short)0, acc, false, false);
    acc = __builtin_amdgcn_wmma_f32_16x16x32_bf16(false, af1[1], false, asbf(u1), (short)0, acc, false, false);
    acc = __builtin_amdgcn_wmma_f32_16x16x32_bf16(false, af1[2], false, asbf(u2), (short)0, acc, false, false);
    acc = __builtin_amdgcn_wmma_f32_16x16x32_bf16(false, af1[3], false, asbf(u3), (short)0, acc, false, false);
    int n = (nt << 4) + m;
    float sc = g1[n] * BN_INV, bb = b1[n], bt = be1[n];
#pragma unroll
    for (int d = 0; d < 8; ++d){
      float v = sc * (acc[d] + bb) + bt;
      v = v > 0.f ? v : 0.f;
      t[(d + (hi ? 8 : 0)) * 260 + n] = v;
    }
  }

  v16bf af2[8];
#pragma unroll
  for (int kc = 0; kc < 8; ++kc){
    int k0 = (kc << 5) + (hi << 3);
    const float* tr = t + m * 260;
    float4 p0 = *(const float4*)(tr + k0);
    float4 p1 = *(const float4*)(tr + k0 + 4);
    float4 p2 = *(const float4*)(tr + k0 + 16);
    float4 p3 = *(const float4*)(tr + k0 + 20);
    cvt8(af2[kc], 0, p0, p1);
    cvt8(af2[kc], 8, p2, p3);
  }

  // matmul2: [16x256] @ [256x128]  (w2 packed kc-major, nt_cnt = 8)
#pragma unroll
  for (int nt = 0; nt < 8; ++nt){
    const u16x16* bp = bfrag_base(w2p, nt, lane);     // kc stride = 8 frags
    u16x16 u0 = bp[0],    u1 = bp[256],  u2 = bp[512],  u3 = bp[768];
    u16x16 u4 = bp[1024], u5 = bp[1280], u6 = bp[1536], u7 = bp[1792];
    v8f acc = v8f_zero();
    acc = __builtin_amdgcn_wmma_f32_16x16x32_bf16(false, af2[0], false, asbf(u0), (short)0, acc, false, false);
    acc = __builtin_amdgcn_wmma_f32_16x16x32_bf16(false, af2[1], false, asbf(u1), (short)0, acc, false, false);
    acc = __builtin_amdgcn_wmma_f32_16x16x32_bf16(false, af2[2], false, asbf(u2), (short)0, acc, false, false);
    acc = __builtin_amdgcn_wmma_f32_16x16x32_bf16(false, af2[3], false, asbf(u3), (short)0, acc, false, false);
    acc = __builtin_amdgcn_wmma_f32_16x16x32_bf16(false, af2[4], false, asbf(u4), (short)0, acc, false, false);
    acc = __builtin_amdgcn_wmma_f32_16x16x32_bf16(false, af2[5], false, asbf(u5), (short)0, acc, false, false);
    acc = __builtin_amdgcn_wmma_f32_16x16x32_bf16(false, af2[6], false, asbf(u6), (short)0, acc, false, false);
    acc = __builtin_amdgcn_wmma_f32_16x16x32_bf16(false, af2[7], false, asbf(u7), (short)0, acc, false, false);
    int n = (nt << 4) + m;
    float sc = g2[n] * BN_INV, bb = b2[n], bt = be2[n];
#pragma unroll
    for (int d = 0; d < 8; ++d){
      int nd = nb * 16 + d + (hi ? 8 : 0);
      if (nd < N){
        float v = sc * (acc[d] + bb) + bt;
        if (!lastlayer) v = v > 0.f ? v : 0.f;
        h[(size_t)nd * 128 + n] = v;
      }
    }
  }
}

__global__ __launch_bounds__(256) void pool_add_kernel(
    const float* __restrict__ h, const int* __restrict__ batch,
    float* __restrict__ pooled, float* __restrict__ cnt, int N)
{
  long long gid = (long long)blockIdx.x * 256 + threadIdx.x;
  int n = (int)(gid >> 5);
  if (n >= N) return;
  int c = (int)(gid & 31) * 4;
  int g = batch[n];
  const float* s = h + (size_t)n * 128 + c;
  float* d = pooled + (size_t)g * 128 + c;
  atomicAdd(d + 0, s[0]); atomicAdd(d + 1, s[1]);
  atomicAdd(d + 2, s[2]); atomicAdd(d + 3, s[3]);
  if (c == 0) atomicAdd(cnt + g, 1.0f);
}

__global__ __launch_bounds__(256) void pool_fin_kernel(float* __restrict__ pooled,
                                                       const float* __restrict__ cnt, int G)
{
  int i = blockIdx.x * 256 + threadIdx.x;
  if (i >= G * 128) return;
  pooled[i] = pooled[i] / (cnt[i >> 7] + 1e-9f);
}

__global__ __launch_bounds__(256) void sab_attn_kernel(
    const float* __restrict__ Q, const float* __restrict__ Kb,
    const float* __restrict__ Vb, float* __restrict__ O, int S)
{
  __shared__ float sc[512];
  __shared__ float red[256];
  int q  = blockIdx.x;
  int hd = blockIdx.y;
  int tid = threadIdx.x;
  const float* Qr = Q + (size_t)q * 128 + hd * 64;
  float lmax = -3.0e38f;
  for (int k = tid; k < S; k += 256){
    const float* Kr = Kb + (size_t)k * 128 + hd * 64;
    float d = 0.f;
    for (int j = 0; j < 64; ++j) d += Qr[j] * Kr[j];
    d *= 0.125f;
    sc[k] = d;
    lmax = fmaxf(lmax, d);
  }
  red[tid] = lmax; __syncthreads();
  for (int s2 = 128; s2 > 0; s2 >>= 1){ if (tid < s2) red[tid] = fmaxf(red[tid], red[tid+s2]); __syncthreads(); }
  float mx = red[0]; __syncthreads();
  float lsum = 0.f;
  for (int k = tid; k < S; k += 256){ float e = __expf(sc[k] - mx); sc[k] = e; lsum += e; }
  red[tid] = lsum; __syncthreads();
  for (int s2 = 128; s2 > 0; s2 >>= 1){ if (tid < s2) red[tid] += red[tid+s2]; __syncthreads(); }
  float inv = 1.0f / red[0];
  if (tid < 64){
    float acc = 0.f;
    for (int k = 0; k < S; ++k) acc += sc[k] * Vb[(size_t)k * 128 + hd * 64 + tid];
    O[(size_t)q * 128 + hd * 64 + tid] = Qr[tid] + acc * inv;
  }
}

__global__ __launch_bounds__(256) void masked_attn_kernel(
    const float* __restrict__ Q, const float* __restrict__ K2,
    const int* __restrict__ mask, float* __restrict__ attn, int S)
{
  __shared__ float sc[512];
  __shared__ float red[256];
  int d   = blockIdx.x;
  int tid = threadIdx.x;
  const float* Qr = Q + (size_t)d * 128;
  float lmax = -3.0e38f;
  for (int s = tid; s < S; s += 256){
    float v;
    if (mask[(size_t)d * S + s]){
      v = NEGV;
    } else {
      const float* Kr = K2 + (size_t)s * 128;
      float acc = 0.f;
      for (int j = 0; j < 128; ++j) acc += Qr[j] * Kr[j];
      v = acc * 0.08838834764831843f;
    }
    sc[s] = v;
    lmax = fmaxf(lmax, v);
  }
  red[tid] = lmax; __syncthreads();
  for (int s2 = 128; s2 > 0; s2 >>= 1){ if (tid < s2) red[tid] = fmaxf(red[tid], red[tid+s2]); __syncthreads(); }
  float mx = red[0]; __syncthreads();
  float lsum = 0.f;
  for (int s = tid; s < S; s += 256){ float e = __expf(sc[s] - mx); sc[s] = e; lsum += e; }
  red[tid] = lsum; __syncthreads();
  for (int s2 = 128; s2 > 0; s2 >>= 1){ if (tid < s2) red[tid] += red[tid+s2]; __syncthreads(); }
  float inv = 1.0f / red[0];
  for (int s = tid; s < S; s += 256) attn[(size_t)d * S + s] = sc[s] * inv;
}

__global__ __launch_bounds__(128) void layernorm_kernel(
    const float* __restrict__ X, const float* __restrict__ g,
    const float* __restrict__ b, float* __restrict__ Y, int R)
{
  int row = blockIdx.x * 4 + (threadIdx.x >> 5);
  if (row >= R) return;
  int lane = threadIdx.x & 31;
  const float* x = X + (size_t)row * 128 + lane * 4;
  float v0 = x[0], v1 = x[1], v2 = x[2], v3 = x[3];
  float s = v0 + v1 + v2 + v3;
#pragma unroll
  for (int o = 16; o > 0; o >>= 1) s += __shfl_xor(s, o, 32);
  float mu = s * (1.0f / 128.0f);
  float d0 = v0-mu, d1 = v1-mu, d2 = v2-mu, d3 = v3-mu;
  float q = d0*d0 + d1*d1 + d2*d2 + d3*d3;
#pragma unroll
  for (int o = 16; o > 0; o >>= 1) q += __shfl_xor(q, o, 32);
  float inv = rsqrtf(q * (1.0f / 128.0f) + 1e-5f);
  int c = lane * 4;
  float* y = Y + (size_t)row * 128 + c;
  y[0] = g[c+0]*d0*inv + b[c+0];
  y[1] = g[c+1]*d1*inv + b[c+1];
  y[2] = g[c+2]*d2*inv + b[c+2];
  y[3] = g[c+3]*d3*inv + b[c+3];
}

__global__ __launch_bounds__(256) void build_aprime_kernel(
    const float* __restrict__ attn, const float* __restrict__ pat,
    float* __restrict__ Ap, int S, long long total)
{
  long long i = (long long)blockIdx.x * 256 + threadIdx.x;
  if (i >= total) return;
  int bs = (int)(i / ((long long)200 * S));
  long long r = i - (long long)bs * 200 * S;
  int d = (int)(r / S);
  int s = (int)(r - (long long)d * S);
  Ap[i] = attn[(size_t)d * S + s] * pat[(size_t)bs * S + s];
}

__global__ __launch_bounds__(128) void final_score_kernel(
    const float* __restrict__ t, const float* __restrict__ w2,
    const float* __restrict__ b2, float* __restrict__ out, int R)
{
  int row = blockIdx.x * 4 + (threadIdx.x >> 5);
  if (row >= R) return;
  int lane = threadIdx.x & 31;
  float2 a = ((const float2*)(t + (size_t)row * 64))[lane];
  float2 w = ((const float2*)w2)[lane];
  float s = a.x * w.x + a.y * w.y;
#pragma unroll
  for (int o = 16; o > 0; o >>= 1) s += __shfl_xor(s, o, 32);
  if (lane == 0){
    float z = s + b2[0];
    out[row] = 1.0f / (1.0f + __expf(-z));
  }
}

// ================================================================ host

extern "C" void kernel_launch(void* const* d_in, const int* in_sizes, int n_in,
                              void* d_out, int out_size, void* d_ws, size_t ws_size,
                              hipStream_t stream) {
  const float* atom_emb  = (const float*)d_in[0];
  const float* bond_emb  = (const float*)d_in[1];
  const float* gin_eps   = (const float*)d_in[2];
  const float* gin_w1    = (const float*)d_in[3];
  const float* gin_b1    = (const float*)d_in[4];
  const float* gin_bn1_g = (const float*)d_in[5];
  const float* gin_bn1_b = (const float*)d_in[6];
  const float* gin_w2    = (const float*)d_in[7];
  const float* gin_b2    = (const float*)d_in[8];
  const float* gin_bn2_g = (const float*)d_in[9];
  const float* gin_bn2_b = (const float*)d_in[10];
  const float* sab_wq    = (const float*)d_in[11];
  const float* sab_wk    = (const float*)d_in[12];
  const float* sab_wv    = (const float*)d_in[13];
  const float* sab_wo    = (const float*)d_in[14];
  const float* sab_bq    = (const float*)d_in[15];
  const float* sab_bk    = (const float*)d_in[16];
  const float* sab_bv    = (const float*)d_in[17];
  const float* sab_bo    = (const float*)d_in[18];
  const float* sab_ln1_b = (const float*)d_in[19];
  const float* sab_ln2_b = (const float*)d_in[20];
  const float* sab_ln1_g = (const float*)d_in[21];
  const float* sab_ln2_g = (const float*)d_in[22];
  const float* agg_wq    = (const float*)d_in[23];
  const float* agg_bq    = (const float*)d_in[24];
  const float* agg_wk    = (const float*)d_in[25];
  const float* agg_bk    = (const float*)d_in[26];
  const float* score_w1  = (const float*)d_in[27];
  const float* score_b1  = (const float*)d_in[28];
  const float* score_w2  = (const float*)d_in[29];
  const float* score_b2  = (const float*)d_in[30];
  const float* patient   = (const float*)d_in[31];
  const float* avg_proj  = (const float*)d_in[32];
  const int*   sub_x     = (const int*)d_in[33];
  const int*   sub_eattr = (const int*)d_in[34];
  const int*   sub_eidx  = (const int*)d_in[35];
  const int*   sub_batch = (const int*)d_in[36];
  const int*   mol_x     = (const int*)d_in[37];
  const int*   mol_eattr = (const int*)d_in[38];
  const int*   mol_eidx  = (const int*)d_in[39];
  const int*   mol_batch = (const int*)d_in[40];
  const int*   maskp     = (const int*)d_in[41];

  const int NS = in_sizes[36];
  const int ES = in_sizes[35] / 2;
  const int NM = in_sizes[40];
  const int EM = in_sizes[39] / 2;
  const int Bn = 64, Dn = 200;
  const int S  = in_sizes[31] / Bn;
  const int Mg = in_sizes[32] / Dn;
  const int Nmax = (NM > NS) ? NM : NS;
  const int BD = Bn * Dn;

  char* wsb = (char*)d_ws;
  size_t off = 0;
  auto alloc = [&](size_t bytes)->char*{
    char* p = wsb + off;
    off += (bytes + 255) & ~(size_t)255;
    return p;
  };
  float* hbuf   = (float*)alloc((size_t)Nmax * 128 * 4);
  float* aggbuf = (float*)alloc((size_t)Nmax * 128 * 4);
  unsigned short* w1p  = (unsigned short*)alloc((size_t)8 * 32768 * 2);
  unsigned short* w2p  = (unsigned short*)alloc((size_t)8 * 32768 * 2);
  unsigned short* sabp = (unsigned short*)alloc((size_t)4 * 16384 * 2);
  unsigned short* aggp = (unsigned short*)alloc((size_t)2 * 16384 * 2);
  unsigned short* sw1p = (unsigned short*)alloc((size_t)8192 * 2);
  int kcM = (Mg + 31) / 32;
  int kcS = (S  + 31) / 32;
  unsigned short* molp  = (unsigned short*)alloc((size_t)8 * kcM * 512 * 2);
  unsigned short* subfp = (unsigned short*)alloc((size_t)8 * kcS * 512 * 2);
  float* pooled_sub = (float*)alloc((size_t)S  * 128 * 4);
  float* cnt_sub    = (float*)alloc((size_t)S  * 4);
  float* pooled_mol = (float*)alloc((size_t)Mg * 128 * 4);
  float* cnt_mol    = (float*)alloc((size_t)Mg * 4);
  float* Qs_   = (float*)alloc((size_t)S * 128 * 4);
  float* Ks_   = (float*)alloc((size_t)S * 128 * 4);
  float* Vs_   = (float*)alloc((size_t)S * 128 * 4);
  float* Ob_   = (float*)alloc((size_t)S * 128 * 4);
  float* L1_   = (float*)alloc((size_t)S * 128 * 4);
  float* L2_   = (float*)alloc((size_t)S * 128 * 4);
  float* subf_ = (float*)alloc((size_t)S * 128 * 4);
  float* molemb_ = (float*)alloc((size_t)Dn * 128 * 4);
  float* Qa_     = (float*)alloc((size_t)Dn * 128 * 4);
  float* K2_     = (float*)alloc((size_t)S  * 128 * 4);
  float* attn_   = (float*)alloc((size_t)Dn * S * 4);
  float* Ap_     = (float*)alloc((size_t)BD * S * 4);
  float* aggo_   = (float*)alloc((size_t)BD * 128 * 4);
  float* tb_     = (float*)alloc((size_t)BD * 64 * 4);
  (void)ws_size; (void)n_in; (void)out_size;

  auto pack = [&](const float* Bm, unsigned short* Pm, int K, int N){
    int kc = (K + 31) / 32, nt = N / 16;
    int total = kc * nt * 512;
    pack_b_kernel<<<(total + 255) / 256, 256, 0, stream>>>(Bm, Pm, K, N, nt, total);
  };
  auto gemm = [&](const float* A, const unsigned short* Bp, const float* bias, float* C,
                  int M, int N, int K, int relu){
    int kc = (K + 31) / 32;
    int tiles = ((M + 15) / 16) * (N / 64);
    gemm_bf16_kernel<<<(tiles + 7) / 8, 256, 0, stream>>>(A, Bp, bias, C, M, N, K, kc, relu);
  };

  for (int e = 0; e < 2; ++e)
    for (int l = 0; l < 4; ++l){
      pack(gin_w1 + (size_t)(e*4+l) * 128 * 256, w1p + (size_t)(e*4+l) * 32768, 128, 256);
      pack(gin_w2 + (size_t)(e*4+l) * 256 * 128, w2p + (size_t)(e*4+l) * 32768, 256, 128);
    }
  pack(sab_wq, sabp + 0,      128, 128);
  pack(sab_wk, sabp + 16384,  128, 128);
  pack(sab_wv, sabp + 32768,  128, 128);
  pack(sab_wo, sabp + 49152,  128, 128);
  pack(agg_wq, aggp + 0,      128, 128);
  pack(agg_wk, aggp + 16384,  128, 128);
  pack(score_w1, sw1p, 128, 64);

  auto run_gin = [&](int enc, const int* x, const int* eattr, const int* eidx,
                     const int* batch, int N, int E, int G,
                     float* pooled, float* cnt){
    long long tot = (long long)N * 32;
    node_embed_kernel<<<(int)((tot + 255) / 256), 256, 0, stream>>>(
        x, atom_emb + (size_t)enc * 9 * 64 * 128, hbuf, N);
    for (int l = 0; l < 4; ++l){
      hipMemsetAsync(aggbuf, 0, (size_t)N * 128 * sizeof(float), stream);
      long long te = (long long)E * 32;
      edge_msg_kernel<<<(int)((te + 255) / 256), 256, 0, stream>>>(
          eidx, eattr, bond_emb + (size_t)(enc*4+l) * 3 * 64 * 128, hbuf, aggbuf, E);
      int ntile = (N + 15) / 16;
      gin_mlp_kernel<<<(ntile + 1) / 2, 64, 0, stream>>>(
          hbuf, aggbuf,
          w1p + (size_t)(enc*4+l) * 32768, w2p + (size_t)(enc*4+l) * 32768,
          gin_b1 + (enc*4+l)*256, gin_bn1_g + (enc*4+l)*256, gin_bn1_b + (enc*4+l)*256,
          gin_b2 + (enc*4+l)*128, gin_bn2_g + (enc*4+l)*128, gin_bn2_b + (enc*4+l)*128,
          gin_eps + (enc*4+l), N, (l == 3) ? 1 : 0);
    }
    hipMemsetAsync(pooled, 0, (size_t)G * 128 * sizeof(float), stream);
    hipMemsetAsync(cnt,    0, (size_t)G * sizeof(float), stream);
    pool_add_kernel<<<(int)((tot + 255) / 256), 256, 0, stream>>>(hbuf, batch, pooled, cnt, N);
    pool_fin_kernel<<<(G * 128 + 255) / 256, 256, 0, stream>>>(pooled, cnt, G);
  };

  run_gin(0, sub_x, sub_eattr, sub_eidx, sub_batch, NS, ES, S, pooled_sub, cnt_sub);
  gemm(pooled_sub, sabp + 0,     sab_bq, Qs_, S, 128, 128, 0);
  gemm(pooled_sub, sabp + 16384, sab_bk, Ks_, S, 128, 128, 0);
  gemm(pooled_sub, sabp + 32768, sab_bv, Vs_, S, 128, 128, 0);
  {
    dim3 g(S, 2);
    sab_attn_kernel<<<g, 256, 0, stream>>>(Qs_, Ks_, Vs_, Ob_, S);
  }
  layernorm_kernel<<<(S + 3) / 4, 128, 0, stream>>>(Ob_, sab_ln1_g, sab_ln1_b, L1_, S);
  gemm(L1_, sabp + 49152, sab_bo, L2_, S, 128, 128, 0);
  layernorm_kernel<<<(S + 3) / 4, 128, 0, stream>>>(L2_, sab_ln2_g, sab_ln2_b, subf_, S);

  run_gin(1, mol_x, mol_eattr, mol_eidx, mol_batch, NM, EM, Mg, pooled_mol, cnt_mol);
  pack(pooled_mol, molp, Mg, 128);
  gemm(avg_proj, molp, nullptr, molemb_, Dn, 128, Mg, 0);

  gemm(molemb_, aggp + 0,     agg_bq, Qa_, Dn, 128, 128, 0);
  gemm(subf_,   aggp + 16384, agg_bk, K2_, S,  128, 128, 0);
  masked_attn_kernel<<<Dn, 256, 0, stream>>>(Qa_, K2_, maskp, attn_, S);

  {
    long long tot = (long long)BD * S;
    build_aprime_kernel<<<(int)((tot + 255) / 256), 256, 0, stream>>>(attn_, patient, Ap_, S, tot);
  }
  pack(subf_, subfp, S, 128);
  gemm(Ap_,  subfp, nullptr,  aggo_, BD, 128, S,   0);
  gemm(aggo_, sw1p, score_b1, tb_,   BD, 64,  128, 1);
  final_score_kernel<<<BD / 4, 128, 0, stream>>>(tb_, score_w2, score_b2, (float*)d_out, BD);
}
